// Protetype_3642132267612
// MI455X (gfx1250) — matrix operand order repaired and checked by reading, not compiled, not run on hardware
//
#include <hip/hip_runtime.h>

// ---------------- problem constants ----------------
#define NCLS    21
#define PADC    32          // classes padded to 32 for two 16-wide WMMA tiles
#define CDIM    256
#define HW      16384       // 128*128
#define NBATCH  8
#define NPIX    (NBATCH*HW) // 131072
#define KWEAK   16
#define KSTRONG 256
#define NEGV    (-1e30f)
#define EPSV    1e-6f
#define MOM     0.99f

typedef __attribute__((ext_vector_type(16))) _Float16 v16h;
typedef __attribute__((ext_vector_type(8)))  float    v8f;

union AFrag { v16h v; _Float16 e[16]; };
union CFrag { v8f  v; float    e[8];  };

__device__ __forceinline__ unsigned fkey(float f) {
    unsigned u = __float_as_uint(f);
    return (u & 0x80000000u) ? ~u : (u | 0x80000000u);
}
__device__ __forceinline__ float fdecode(unsigned k) {
    unsigned u = (k & 0x80000000u) ? (k & 0x7FFFFFFFu) : ~k;
    return __uint_as_float(u);
}

// gfx1250 async global->LDS (16B per lane, tracked by ASYNCcnt)
__device__ __forceinline__ void async_g2l_b128(const float* g, const float* lds_generic) {
    unsigned la = (unsigned)(unsigned long long)(const void*)lds_generic; // addr[31:0] == LDS offset
    unsigned long long ga = (unsigned long long)(const void*)g;
    asm volatile("global_load_async_to_lds_b128 %0, %1, off" :: "v"(la), "v"(ga) : "memory");
}
__device__ __forceinline__ void wait_async0() {
    asm volatile("s_wait_asynccnt 0" ::: "memory");
}

// ---------------- kernel 1: zero accumulators ----------------
__global__ void init_kernel(_Float16* pnH, float* wsum, float* simSum, int* cntOk) {
    int i = blockIdx.x * blockDim.x + threadIdx.x;
    if (i < PADC * CDIM) pnH[i] = (_Float16)0.f;
    if (i < NCLS * CDIM) wsum[i] = 0.f;
    if (i == 0) { *simSum = 0.f; *cntOk = 0; }
}

// ---------------- kernel 2: normalize prototypes ----------------
__global__ void proto_kernel(const float* __restrict__ mb, _Float16* __restrict__ pnH) {
    __shared__ float red[256];
    int k = blockIdx.x, t = threadIdx.x;
    float v = mb[k * CDIM + t];
    red[t] = v * v;
    __syncthreads();
    for (int s = 128; s > 0; s >>= 1) {
        if (t < s) red[t] += red[t + s];
        __syncthreads();
    }
    float inv = 1.0f / fmaxf(sqrtf(red[0]), EPSV);
    pnH[k * CDIM + t] = (_Float16)(v * inv);
}

// ---------------- kernel 3: pack per-pixel flags ----------------
__global__ void flags_kernel(const int* __restrict__ pred,
                             const int* __restrict__ ign,
                             const float* __restrict__ prob,
                             unsigned char* __restrict__ flags) {
    int n = blockIdx.x * blockDim.x + threadIdx.x;
    if (n >= NPIX) return;
    int s = pred[n] & 31;
    bool valid = (ign[n] != 255);
    bool conf  = valid && (prob[n] > 0.95f);
    flags[n] = (unsigned char)(s | (conf ? 32 : 0) | (valid ? 64 : 0));
}

// ---------------- kernel 4: masked similarities via async-LDS + WMMA ----------------
// grid = NPIX/128 blocks, 256 threads (8 waves, one 16-pixel tile each).
// Also folds the class-conditional feature sums (EMA numerator) in f32.
#define FROW 132   // padded f32 LDS row stride (floats): 132 banks -> conflict-free
#define PROW 264   // padded prototype row stride (halfs)
__global__ __launch_bounds__(256)
void sim_kernel(const float* __restrict__ fw, const float* __restrict__ fs,
                const _Float16* __restrict__ pnH,
                const unsigned char* __restrict__ flags,
                float* __restrict__ sw, float* __restrict__ ss,
                float* __restrict__ wsum) {
    __shared__ float    ldsWf[32][FROW];
    __shared__ float    ldsSf[32][FROW];
    __shared__ _Float16 ldsPn[PADC * PROW];
    __shared__ unsigned char flLds[128];

    const int t    = threadIdx.x;
    const int lane = t & 31;
    const int wave = t >> 5;
    const int pix0 = blockIdx.x * 128;          // all 128 pixels in same batch image
    const int b    = pix0 >> 14;
    const int hw0  = pix0 & (HW - 1);
    const size_t base = (size_t)b * CDIM * HW + hw0;

    // stage prototypes once (padded rows) + flags for this block's pixels
    for (int i = t; i < PADC * CDIM; i += 256) {
        int kcls = i >> 8, c = i & 255;
        ldsPn[kcls * PROW + c] = pnH[i];
    }
    if (t < 128) flLds[t] = flags[pix0 + t];

    CFrag aw0, aw1, as0, as1;
    for (int i = 0; i < 8; ++i) { aw0.e[i] = 0.f; aw1.e[i] = 0.f; as0.e[i] = 0.f; as1.e[i] = 0.f; }
    float sqW = 0.f, sqS = 0.f;

    const int m   = lane & 15;
    const int kb  = (lane >> 4) * 8;
    const int kk0 = (lane >> 4) * 16;
    const int col = lane & 15;

    for (int kc = 0; kc < 8; ++kc) {
        // async DMA this chunk (32 channels x 128 px, both tensors) into LDS
#pragma unroll
        for (int i = 0; i < 4; ++i) {
            int slot = t + i * 256;             // 1024 x 16B per tensor
            int cl = slot >> 5, p4 = (slot & 31) * 4;
            size_t g = base + (size_t)(kc * 32 + cl) * HW + p4;
            async_g2l_b128(fw + g, &ldsWf[cl][p4]);
            async_g2l_b128(fs + g, &ldsSf[cl][p4]);
        }
        if (kc < 7) {   // L2 prefetch of next chunk
            size_t pa = base + (size_t)((kc + 1) * 32) * HW + (t & 127);
            __builtin_prefetch(&fw[pa], 0, 1);
            __builtin_prefetch(&fs[pa], 0, 1);
        }
        wait_async0();
        __syncthreads();

        AFrag aW, aS, b0, b1;
#pragma unroll
        for (int e = 0; e < 16; ++e) {
            int cl = kk0 + e;                               // contiguous k order (matches B)
            aW.e[e] = (_Float16)ldsWf[cl][wave * 16 + m];
            aS.e[e] = (_Float16)ldsSf[cl][wave * 16 + m];
            int kkk = kc * 32 + kk0 + e;                    // B-matrix layout
            b0.e[e] = ldsPn[col * PROW + kkk];
            b1.e[e] = ldsPn[(col + 16) * PROW + kkk];
        }
        aw0.v = __builtin_amdgcn_wmma_f32_16x16x32_f16(false, aW.v, false, b0.v, (short)0, aw0.v, false, false);
        aw1.v = __builtin_amdgcn_wmma_f32_16x16x32_f16(false, aW.v, false, b1.v, (short)0, aw1.v, false, false);
        as0.v = __builtin_amdgcn_wmma_f32_16x16x32_f16(false, aS.v, false, b0.v, (short)0, as0.v, false, false);
        as1.v = __builtin_amdgcn_wmma_f32_16x16x32_f16(false, aS.v, false, b1.v, (short)0, as1.v, false, false);

        {   // per-pixel sum of squares in f32 (lane covers pixel lane&15, half of K)
            int p = lane & 15, h = lane >> 4;
#pragma unroll
            for (int q = 0; q < 16; ++q) {
                float xw = ldsWf[h * 16 + q][wave * 16 + p];
                float xs = ldsSf[h * 16 + q][wave * 16 + p];
                sqW += xw * xw;
                sqS += xs * xs;
            }
        }
        {   // folded class-conditional sum (EMA numerator), exact f32
#pragma unroll
            for (int i = 0; i < 4; ++i) {
                int slot = t + i * 256;
                int cl = slot >> 5, p4 = (slot & 31) * 4;
                int c = kc * 32 + cl;
#pragma unroll
                for (int q = 0; q < 4; ++q) {
                    unsigned f = flLds[p4 + q];
                    if (f & 32) atomicAdd(&wsum[(f & 31) * CDIM + c], ldsWf[cl][p4 + q]);
                }
            }
        }
        __syncthreads();
    }

    sqW += __shfl_xor(sqW, 16, 32);
    sqS += __shfl_xor(sqS, 16, 32);
    float invW = 1.0f / fmaxf(sqrtf(sqW), EPSV);
    float invS = 1.0f / fmaxf(sqrtf(sqS), EPSV);

#pragma unroll
    for (int v = 0; v < 8; ++v) {
        int mm = v + ((lane >> 4) << 3);
        float iw = __shfl(invW, mm, 32);
        float is = __shfl(invS, mm, 32);
        int n = pix0 + wave * 16 + mm;
        unsigned f = flLds[wave * 16 + mm];
        int seg = f & 31;
        bool conf = (f & 32) != 0, val = (f & 64) != 0;
        int k0 = col;
        sw[(size_t)k0 * NPIX + n] = (seg == k0 && conf) ? aw0.e[v] * iw : NEGV;
        ss[(size_t)k0 * NPIX + n] = (seg == k0 && val)  ? -(as0.e[v] * is) : NEGV;
        int k1 = col + 16;
        if (k1 < NCLS) {
            sw[(size_t)k1 * NPIX + n] = (seg == k1 && conf) ? aw1.e[v] * iw : NEGV;
            ss[(size_t)k1 * NPIX + n] = (seg == k1 && val)  ? -(as1.e[v] * is) : NEGV;
        }
    }
}

// ---------------- kernel 5: per-class top-K (histogram + bitonic) ----------------
#define NBINS 4096
#define CAP   1024
__global__ __launch_bounds__(1024)
void topk_kernel(const float* __restrict__ sw, const float* __restrict__ ss,
                 int* __restrict__ idxW, int* __restrict__ idxS,
                 int* __restrict__ nW, int* __restrict__ nS) {
    __shared__ unsigned hist[NBINS];
    __shared__ float cv[CAP];
    __shared__ int   ci[CAP];
    __shared__ int sThr, sCnum, sTotal;

    int t = threadIdx.x;
    int k = blockIdx.x % NCLS;
    bool strong = blockIdx.x >= NCLS;
    const float* arr = (strong ? ss : sw) + (size_t)k * NPIX;
    int K = strong ? KSTRONG : KWEAK;
    int* idxo = strong ? (idxS + k * KSTRONG) : (idxW + k * KWEAK);
    int* cnto = strong ? (nS + k) : (nW + k);

    for (int i = t; i < NBINS; i += 1024) hist[i] = 0u;
    if (t == 0) { sTotal = 0; sCnum = 0; }
    __syncthreads();

    int local = 0;
    for (int i = t; i < NPIX; i += 1024) {
        float v = arr[i];
        if (v > -1e29f) {
            int bb = (int)((v + 1.0f) * (NBINS / 2));
            bb = min(max(bb, 0), NBINS - 1);
            atomicAdd(&hist[bb], 1u);
            local++;
        }
    }
    atomicAdd(&sTotal, local);
    __syncthreads();

    if (t == 0) {
        int cum = 0, thr = 0;
        for (int bb = NBINS - 1; bb >= 0; --bb) {
            cum += (int)hist[bb];
            if (cum >= K) { thr = bb; break; }
        }
        sThr = thr;
    }
    __syncthreads();
    int thr = sThr;

    for (int i = t; i < NPIX; i += 1024) {
        float v = arr[i];
        if (v > -1e29f) {
            int bb = (int)((v + 1.0f) * (NBINS / 2));
            bb = min(max(bb, 0), NBINS - 1);
            if (bb >= thr) {
                int pos = atomicAdd(&sCnum, 1);
                if (pos < CAP) { cv[pos] = v; ci[pos] = i; }
            }
        }
    }
    __syncthreads();
    int cn = min(sCnum, CAP);
    if (t >= cn) { cv[t] = NEGV; ci[t] = -1; }
    __syncthreads();

    for (int sz = 2; sz <= CAP; sz <<= 1) {
        for (int j = sz >> 1; j > 0; j >>= 1) {
            int i = t, ixj = i ^ j;
            if (ixj > i) {
                bool desc = ((i & sz) == 0);
                float vi = cv[i], vj = cv[ixj];
                if ((vi < vj) == desc) {
                    cv[i] = vj; cv[ixj] = vi;
                    int tmp = ci[i]; ci[i] = ci[ixj]; ci[ixj] = tmp;
                }
            }
            __syncthreads();
        }
    }
    if (t < K) idxo[t] = ci[t];
    if (t == 0) *cnto = sTotal;
}

// ---------------- kernel 6: per-class contrastive term via WMMA ----------------
// grid = 21 blocks, 128 threads (4 waves)
#define SROW 264   // padded LDS row stride (halfs)
__global__ __launch_bounds__(128)
void loss_kernel(const float* __restrict__ fw, const float* __restrict__ fs,
                 const int* __restrict__ idxW, const int* __restrict__ idxS,
                 const int* __restrict__ nW, const int* __restrict__ nS,
                 float* __restrict__ simSum, int* __restrict__ cntOk) {
    __shared__ _Float16 Wn[16][SROW];
    __shared__ _Float16 Sn[64][SROW];
    __shared__ float wSq[16], sSq[64], wInv[16], sInv[64];
    __shared__ unsigned mKey[16];

    int t = threadIdx.x, lane = t & 31, wave = t >> 5;
    int k = blockIdx.x;
    int nw = nW[k], ns = nS[k];
    if (nw <= 0 || ns <= 0) return;              // class_ok false
    int kw = min(KWEAK, nw), ks = min(KSTRONG, ns);

    if (t < 16) { wSq[t] = 0.f; mKey[t] = fkey(NEGV); }
    __syncthreads();

    for (int w = 0; w < 16; ++w) {               // ranks = max(w-1, 0)
        int r = (w == 0) ? 0 : (w - 1);
        int src = (w < kw) ? idxW[k * KWEAK + r] : -1;
        for (int c = t; c < CDIM; c += 128) {
            float x = 0.f;
            if (src >= 0) {
                int b = src >> 14, hw = src & (HW - 1);
                x = fw[((size_t)b * CDIM + c) * HW + hw];
            }
            Wn[w][c] = (_Float16)x;
            atomicAdd(&wSq[w], x * x);
        }
    }
    __syncthreads();
    if (t < 16) wInv[t] = 1.0f / fmaxf(sqrtf(wSq[t]), EPSV);

    const int m   = lane & 15;
    const int kb  = (lane >> 4) * 8;
    const int kk0 = (lane >> 4) * 16;

    for (int chunk = 0; chunk < 4; ++chunk) {
        __syncthreads();
        if (t < 64) sSq[t] = 0.f;
        __syncthreads();
        for (int j = 0; j < 64; ++j) {
            int row = chunk * 64 + j;
            int src = (row < ks) ? idxS[k * KSTRONG + row] : -1;
            for (int c = t; c < CDIM; c += 128) {
                float x = 0.f;
                if (src >= 0) {
                    int b = src >> 14, hw = src & (HW - 1);
                    x = fs[((size_t)b * CDIM + c) * HW + hw];
                }
                Sn[j][c] = (_Float16)x;
                atomicAdd(&sSq[j], x * x);
            }
        }
        __syncthreads();
        if (t < 64) sInv[t] = 1.0f / fmaxf(sqrtf(sSq[t]), EPSV);
        __syncthreads();

        CFrag acc;
        for (int i = 0; i < 8; ++i) acc.e[i] = 0.f;
        for (int kc = 0; kc < 8; ++kc) {
            AFrag a, bf;
#pragma unroll
            for (int e = 0; e < 16; ++e) {
                int cl = kk0 + e;
                a.e[e]  = Wn[m][kc * 32 + cl];
                bf.e[e] = Sn[wave * 16 + m][kc * 32 + kk0 + e];
            }
            acc.v = __builtin_amdgcn_wmma_f32_16x16x32_f16(false, a.v, false, bf.v, (short)0, acc.v, false, false);
        }
#pragma unroll
        for (int v = 0; v < 8; ++v) {
            int mm = v + ((lane >> 4) << 3);
            int nn = lane & 15;
            int srow = chunk * 64 + wave * 16 + nn;
            float cos = (srow < ks) ? acc.e[v] * wInv[mm] * sInv[wave * 16 + nn] : NEGV;
            for (int off = 1; off < 16; off <<= 1)
                cos = fmaxf(cos, __shfl_xor(cos, off, 32));
            if ((lane & 15) == 0) atomicMax(&mKey[mm], fkey(cos));
        }
    }
    __syncthreads();
    if (t == 0) {
        float s = 0.f;
        for (int w = 0; w < kw; ++w) s += fdecode(mKey[w]);
        atomicAdd(simSum, s);
        atomicAdd(cntOk, ks);
    }
}

// ---------------- kernel 7: finalize loss + memory-bank EMA ----------------
__global__ void final_kernel(const float* __restrict__ mb,
                             const float* __restrict__ wsum,
                             const int* __restrict__ nW,
                             const float* __restrict__ simSum,
                             const int* __restrict__ cntOk,
                             float* __restrict__ out) {
    int k = blockIdx.x, c = threadIdx.x;
    int nw = nW[k];
    float p = mb[k * CDIM + c];
    float o = p;
    if (nw > 0) {
        float mean = wsum[k * CDIM + c] / (float)nw;
        o = MOM * p + (1.0f - MOM) * mean;
    }
    out[1 + k * CDIM + c] = o;
    if (k == 0 && c == 0) {
        int ct = *cntOk;
        out[0] = (ct > 0) ? (1.0f - (*simSum) / (float)ct) : 0.0f;
    }
}

// ---------------- host launcher ----------------
extern "C" void kernel_launch(void* const* d_in, const int* in_sizes, int n_in,
                              void* d_out, int out_size, void* d_ws, size_t ws_size,
                              hipStream_t stream) {
    (void)in_sizes; (void)n_in; (void)out_size; (void)ws_size;
    const float* fw   = (const float*)d_in[0];
    const float* fs   = (const float*)d_in[1];
    const float* prob = (const float*)d_in[2];
    const float* mb   = (const float*)d_in[3];
    const int*   pred = (const int*)d_in[4];
    const int*   ign  = (const int*)d_in[5];
    float* out = (float*)d_out;

    char* ws = (char*)d_ws;
    size_t off = 0;
    auto wsalloc = [&](size_t bytes) -> void* {
        void* p = ws + off;
        off += (bytes + 255) & ~(size_t)255;
        return p;
    };
    _Float16*      pnH    = (_Float16*)     wsalloc((size_t)PADC * CDIM * 2);
    float*         sw     = (float*)        wsalloc((size_t)NCLS * NPIX * 4);
    float*         ss     = (float*)        wsalloc((size_t)NCLS * NPIX * 4);
    int*           idxW   = (int*)          wsalloc((size_t)NCLS * KWEAK * 4);
    int*           idxS   = (int*)          wsalloc((size_t)NCLS * KSTRONG * 4);
    int*           nW     = (int*)          wsalloc(NCLS * 4);
    int*           nS     = (int*)          wsalloc(NCLS * 4);
    float*         wsum   = (float*)        wsalloc((size_t)NCLS * CDIM * 4);
    unsigned char* flags  = (unsigned char*)wsalloc(NPIX);
    float*         simSum = (float*)        wsalloc(4);
    int*           cntOk  = (int*)          wsalloc(4);

    init_kernel<<<(PADC * CDIM + 255) / 256, 256, 0, stream>>>(pnH, wsum, simSum, cntOk);
    proto_kernel<<<NCLS, 256, 0, stream>>>(mb, pnH);
    flags_kernel<<<NPIX / 256, 256, 0, stream>>>(pred, ign, prob, flags);
    sim_kernel<<<NPIX / 128, 256, 0, stream>>>(fw, fs, pnH, flags, sw, ss, wsum);
    topk_kernel<<<2 * NCLS, 1024, 0, stream>>>(sw, ss, idxW, idxS, nW, nS);
    loss_kernel<<<NCLS, 128, 0, stream>>>(fw, fs, idxW, idxS, nW, nS, simSum, cntOk);
    final_kernel<<<NCLS, CDIM, 0, stream>>>(mb, wsum, nW, simSum, cntOk, out);
}
